// HeteroGAT_56427280335129
// MI455X (gfx1250) — matrix operand order repaired and checked by reading, not compiled, not run on hardware
//
#include <hip/hip_runtime.h>
#include <hip/hip_bf16.h>
#include <math.h>

// ---------------------------------------------------------------------------
// HeteroGAT (2-layer, 3-relation) for MI455X / gfx1250.
//  - Dense transforms: v_wmma_f32_16x16x32_bf16, A-strip held in registers
//    (one streaming pass over X), W tiles staged in LDS (padded vs bank
//    conflicts), compile-time Dout via templates.
//  - Edge softmax + scatter: hardware f32 global atomics; gather working sets
//    (hs 51MB + accumulator 51MB) are L2-resident on the 192MB L2.
// ---------------------------------------------------------------------------

#define N_USERS 50000
#define N_ITEMS 50000
#define D_IN    256
#define E_NUM   400000
#define H1      4
#define C1      64
#define D1      256      // H1*C1
#define H2      1
#define C2      128
#define D2      128
#define NEG_SLOPE 0.2f
#define EPS_F   1e-16f

typedef __attribute__((ext_vector_type(16))) __bf16 v16bf;
typedef __attribute__((ext_vector_type(8)))  float  v8f;

__device__ __forceinline__ unsigned short f2bf_bits(float f) {
  unsigned int u = __float_as_uint(f);
  u += 0x7FFFu + ((u >> 16) & 1u);              // round-to-nearest-even
  return (unsigned short)(u >> 16);
}
__device__ __forceinline__ __bf16 f2bf(float f) {
  return __builtin_bit_cast(__bf16, f2bf_bits(f));
}

__device__ __forceinline__ void atomicAddF(float* p, float v) {
  unsafeAtomicAdd(p, v);                        // global_atomic_add_f32
}

__device__ __forceinline__ void atomicMaxF(float* addr, float val) {
  int* ia = reinterpret_cast<int*>(addr);
  int old = __float_as_int(-__builtin_huge_valf());
  while (__int_as_float(old) < val) {
    int assumed = old;
    old = atomicCAS(ia, assumed, __float_as_int(val));
    if (old == assumed) break;
  }
}

// ---------------------------------------------------------------------------
// GEMM: Out[N,DOUT] = X[N,256] @ W[256,DOUT]  (f32 in/out, bf16 WMMA).
// Block = 256 threads (8 waves). Block owns 128 rows x full DOUT.
// Wave owns 16 rows: A strip (K=256) preloaded to 8 bf16 fragments (regs),
// W column tiles staged in LDS, 8 WMMAs per 16-wide column tile.
// ---------------------------------------------------------------------------
template <int DOUT>
__global__ void __launch_bounds__(256)
gemm_wmma_bf16(const float* __restrict__ X, const float* __restrict__ W,
               float* __restrict__ Out, int N) {
  constexpr int K = 256;
  constexpr int LDS_STRIDE = 264;               // 256 + 8 shorts pad (16B)
  __shared__ unsigned short Bt[16 * LDS_STRIDE];

  const int tid     = threadIdx.x;
  const int lane    = tid & 31;
  const int wave    = tid >> 5;
  const int halfSel = lane >> 4;                // 0: lanes 0-15, 1: 16-31
  const int l16     = lane & 15;
  const int rowBase = blockIdx.x * 128 + wave * 16;
  const bool valid  = (rowBase + 16) <= N;      // wave-uniform (N % 16 == 0)

  int arow = rowBase + l16;
  if (arow >= N) arow = N - 1;                  // clamp loads only
  const float* xrow = X + (size_t)arow * K;

  // ---- Preload A strip: 8 fragments covering K=0..255 (ISA 16x32 layout):
  //  lanes 0-15 : elems 0..7 -> K=kb+e,   elems 8..15 -> K=kb+16+e
  //  lanes 16-31: elems 0..7 -> K=kb+8+e, elems 8..15 -> K=kb+24+e
  v16bf areg[8];
#pragma unroll
  for (int ks = 0; ks < 8; ++ks) {
    const int kb = ks * 32;
    const float4* p0 = reinterpret_cast<const float4*>(xrow + kb + halfSel * 8);
    const float4* p1 = reinterpret_cast<const float4*>(xrow + kb + 16 + halfSel * 8);
    float4 f0 = p0[0], f1 = p0[1], f2 = p1[0], f3 = p1[1];
    v16bf a;
    a[0]  = f2bf(f0.x); a[1]  = f2bf(f0.y); a[2]  = f2bf(f0.z); a[3]  = f2bf(f0.w);
    a[4]  = f2bf(f1.x); a[5]  = f2bf(f1.y); a[6]  = f2bf(f1.z); a[7]  = f2bf(f1.w);
    a[8]  = f2bf(f2.x); a[9]  = f2bf(f2.y); a[10] = f2bf(f2.z); a[11] = f2bf(f2.w);
    a[12] = f2bf(f3.x); a[13] = f2bf(f3.y); a[14] = f2bf(f3.z); a[15] = f2bf(f3.w);
    areg[ks] = a;
  }

  const int ldCol = tid & 15;                   // cooperative W loader coords
  const int ldK0  = tid >> 4;                   // 0..15

  for (int nt = 0; nt < DOUT / 16; ++nt) {
    const int n0 = nt * 16;
    __syncthreads();                            // previous tile fully consumed
    // Stage W[:, n0..n0+15] as bf16 into LDS (coalesced: 16 lanes = 64B row)
#pragma unroll
    for (int kk = 0; kk < 16; ++kk) {
      const int k = kk * 16 + ldK0;
      Bt[ldCol * LDS_STRIDE + k] = f2bf_bits(W[(size_t)k * DOUT + n0 + ldCol]);
    }
    __syncthreads();

    v8f acc = {0.f, 0.f, 0.f, 0.f, 0.f, 0.f, 0.f, 0.f};
#pragma unroll
    for (int ks = 0; ks < 8; ++ks) {
      // B fragment (32x16): lane = column l16; elems 0..7 K=kb+8*half+e,
      // elems 8..15 K=kb+16+8*half+(e-8). Two 16B runs per lane -> ds_load_b128.
      const int kOff = ks * 32 + halfSel * 8;
      v16bf b;
#pragma unroll
      for (int e = 0; e < 16; ++e)
        b[e] = __builtin_bit_cast(__bf16, Bt[l16 * LDS_STRIDE + kOff + (e & 7) + (e >> 3) * 16]);
      acc = __builtin_amdgcn_wmma_f32_16x16x32_bf16(false, areg[ks], false, b,
                                                    (short)0, acc, false, false);
    }

    if (valid) {
      // C/D layout: VGPR r -> row = rowBase + r + 8*halfSel, col = n0 + l16
      float* orow = Out + (size_t)(rowBase + 8 * halfSel) * DOUT + n0 + l16;
#pragma unroll
      for (int r = 0; r < 8; ++r)
        orow[(size_t)r * DOUT] = acc[r];
    }
  }
}

// alpha[n,h] = sum_c Hm[n, h*C + c] * a[h*C + c]   (one wave per (n,h))
__global__ void alpha_kernel(const float* __restrict__ Hm,
                             const float* __restrict__ a,
                             float* __restrict__ alpha,
                             int N, int H, int C) {
  int wid  = (blockIdx.x * blockDim.x + threadIdx.x) >> 5;
  int lane = threadIdx.x & 31;
  if (wid >= N * H) return;                     // wave-uniform exit
  int n = wid / H, h = wid % H;
  const float* hp = Hm + (size_t)n * H * C + h * C;
  const float* ap = a + h * C;
  float s = 0.f;
  for (int c = lane; c < C; c += 32) s += hp[c] * ap[c];
#pragma unroll
  for (int off = 16; off > 0; off >>= 1) s += __shfl_down(s, off, 32);
  if (lane == 0) alpha[wid] = s;
}

__global__ void fill_kernel(float* __restrict__ p, float v, unsigned n) {
  unsigned i = blockIdx.x * blockDim.x + threadIdx.x;
  if (i < n) p[i] = v;
}

__global__ void edge_logit_max(const int* __restrict__ src, const int* __restrict__ dst,
                               const float* __restrict__ as_, const float* __restrict__ ad_,
                               float* __restrict__ m, int E, int H) {
  int idx = blockIdx.x * blockDim.x + threadIdx.x;
  if (idx >= E * H) return;
  int e = idx / H, h = idx % H;
  float v = as_[src[e] * H + h] + ad_[dst[e] * H + h];
  v = v > 0.f ? v : NEG_SLOPE * v;
  atomicMaxF(&m[dst[e] * H + h], v);
}

__global__ void edge_expsum(const int* __restrict__ src, const int* __restrict__ dst,
                            const float* __restrict__ as_, const float* __restrict__ ad_,
                            const float* __restrict__ m, float* __restrict__ denom,
                            float* __restrict__ ex, int E, int H) {
  int idx = blockIdx.x * blockDim.x + threadIdx.x;
  if (idx >= E * H) return;
  int e = idx / H, h = idx % H;
  int d = dst[e];
  float v = as_[src[e] * H + h] + ad_[d * H + h];
  v = v > 0.f ? v : NEG_SLOPE * v;
  float mv = m[d * H + h];
  if (mv == -__builtin_huge_valf()) mv = 0.f;
  float x = expf(v - mv);
  ex[idx] = x;
  atomicAddF(&denom[d * H + h], x);
}

// ex[e,h] <- ex[e,h] / (denom[dst,h] + eps)   (becomes the softmax weight)
__global__ void edge_alpha(const int* __restrict__ dst, const float* __restrict__ denom,
                           float* __restrict__ ex, int E, int H) {
  int idx = blockIdx.x * blockDim.x + threadIdx.x;
  if (idx >= E * H) return;
  int e = idx / H, h = idx % H;
  ex[idx] = ex[idx] / (denom[dst[e] * H + h] + EPS_F);
}

// out[dst, c..c+3] += hs[src, c..c+3] * alpha[e, c/C]
template <int DOUT, int C>
__global__ void edge_msg(const int* __restrict__ src, const int* __restrict__ dst,
                         const float* __restrict__ hs, const float* __restrict__ alpha,
                         float* __restrict__ out, int total4) {
  constexpr int V = DOUT / 4;                   // float4s per edge
  constexpr int H = DOUT / C;
  int idx = blockIdx.x * blockDim.x + threadIdx.x;
  if (idx >= total4) return;
  int e = idx / V, q = idx - e * V;
  int c = q * 4;
  int h = c / C;                                // uniform within the float4
  float al = alpha[e * H + h];
  const float4 hv =
      *reinterpret_cast<const float4*>(hs + (size_t)src[e] * DOUT + c);
  float* op = out + (size_t)dst[e] * DOUT + c;
  atomicAddF(op + 0, hv.x * al);
  atomicAddF(op + 1, hv.y * al);
  atomicAddF(op + 2, hv.z * al);
  atomicAddF(op + 3, hv.w * al);
}

// buf = scale*buf + scale*(b1 + b2); optional ELU. In place.
__global__ void bias_act(float* __restrict__ buf, const float* __restrict__ b1,
                         const float* __restrict__ b2, int n, int dm,
                         float scale, int doElu) {
  int idx = blockIdx.x * blockDim.x + threadIdx.x;
  if (idx >= n * dm) return;
  int c = idx % dm;
  float bb = b1[c] + (b2 ? b2[c] : 0.f);
  float v = buf[idx] * scale + scale * bb;
  if (doElu) v = v > 0.f ? v : expm1f(v);
  buf[idx] = v;
}

// ---------------------------------------------------------------------------
static inline unsigned cdiv(unsigned a, unsigned b) { return (a + b - 1) / b; }

template <int H, int C>
static void run_relation(const float* xsrc, const float* xdst, int n_src, int n_dst,
                         const int* edges, const float* Ws, const float* Wd,
                         const float* a_s, const float* a_d,
                         float* hsB, float* hdB, float* alS, float* alD,
                         float* mB, float* denB, float* exB,
                         float* outAcc, hipStream_t stream) {
  constexpr int DOUT = H * C;
  dim3 blk(256);
  gemm_wmma_bf16<DOUT><<<cdiv(n_src, 128), blk, 0, stream>>>(xsrc, Ws, hsB, n_src);
  gemm_wmma_bf16<DOUT><<<cdiv(n_dst, 128), blk, 0, stream>>>(xdst, Wd, hdB, n_dst);
  alpha_kernel<<<cdiv(n_src * H * 32, 256), 256, 0, stream>>>(hsB, a_s, alS, n_src, H, C);
  alpha_kernel<<<cdiv(n_dst * H * 32, 256), 256, 0, stream>>>(hdB, a_d, alD, n_dst, H, C);
  fill_kernel<<<cdiv(n_dst * H, 256), 256, 0, stream>>>(mB, -__builtin_huge_valf(),
                                                        (unsigned)(n_dst * H));
  fill_kernel<<<cdiv(n_dst * H, 256), 256, 0, stream>>>(denB, 0.f, (unsigned)(n_dst * H));
  const int* src = edges;
  const int* dst = edges + E_NUM;
  edge_logit_max<<<cdiv(E_NUM * H, 256), 256, 0, stream>>>(src, dst, alS, alD, mB, E_NUM, H);
  edge_expsum<<<cdiv(E_NUM * H, 256), 256, 0, stream>>>(src, dst, alS, alD, mB, denB, exB,
                                                        E_NUM, H);
  edge_alpha<<<cdiv(E_NUM * H, 256), 256, 0, stream>>>(dst, denB, exB, E_NUM, H);
  const int total4 = E_NUM * (DOUT / 4);
  edge_msg<DOUT, C><<<cdiv(total4, 256), 256, 0, stream>>>(src, dst, hsB, exB, outAcc,
                                                           total4);
}

extern "C" void kernel_launch(void* const* d_in, const int* in_sizes, int n_in,
                              void* d_out, int out_size, void* d_ws, size_t ws_size,
                              hipStream_t stream) {
  (void)in_sizes; (void)n_in; (void)out_size; (void)ws_size;

  const float* xu = (const float*)d_in[0];
  const float* xi = (const float*)d_in[1];
  const int* e_rates = (const int*)d_in[2];
  const int* e_rev   = (const int*)d_in[3];
  const int* e_fol   = (const int*)d_in[4];
  // params pytree flattened depth-first: layer -> relation -> (Ws,Wd,a_s,a_d,b)
  auto P = [&](int layer, int rel, int w) -> const float* {
    return (const float*)d_in[5 + layer * 15 + rel * 5 + w];
  };

  // Workspace layout (floats); total ~53.6M floats (~215 MB)
  float* ws   = (float*)d_ws;
  float* A    = ws;                 // hs buffer           (12.8M)
  float* B    = ws + 12800000u;     // hd buffer           (12.8M)
  float* OUTI = ws + 25600000u;     // item accum -> xi1   (12.8M)
  float* OUTU = ws + 38400000u;     // user accum -> xu1   (12.8M)
  float* ALS  = ws + 51200000u;     // alpha_src           (0.2M)
  float* ALD  = ws + 51400000u;     // alpha_dst           (0.2M)
  float* MB   = ws + 51600000u;     // segment max         (0.2M)
  float* DEN  = ws + 51800000u;     // segment sum         (0.2M)
  float* EX   = ws + 52000000u;     // per-edge exp/alpha  (1.6M)

  // ---------------- Layer 1 (Din=256, H=4, C=64, out 256, concat) ----------
  fill_kernel<<<cdiv(N_ITEMS * D1, 256), 256, 0, stream>>>(OUTI, 0.f, N_ITEMS * D1);
  fill_kernel<<<cdiv(N_USERS * D1, 256), 256, 0, stream>>>(OUTU, 0.f, N_USERS * D1);

  run_relation<H1, C1>(xu, xi, N_USERS, N_ITEMS, e_rates,
                       P(0,0,0), P(0,0,1), P(0,0,2), P(0,0,3),
                       A, B, ALS, ALD, MB, DEN, EX, OUTI, stream);
  run_relation<H1, C1>(xi, xu, N_ITEMS, N_USERS, e_rev,
                       P(0,1,0), P(0,1,1), P(0,1,2), P(0,1,3),
                       A, B, ALS, ALD, MB, DEN, EX, OUTU, stream);
  run_relation<H1, C1>(xu, xu, N_USERS, N_USERS, e_fol,
                       P(0,2,0), P(0,2,1), P(0,2,2), P(0,2,3),
                       A, B, ALS, ALD, MB, DEN, EX, OUTU, stream);

  // xi1 = elu(out_i + b_rates) ; xu1 = elu(0.5*(s_rev+s_fol) + 0.5*(b_rev+b_fol))
  bias_act<<<cdiv(N_ITEMS * D1, 256), 256, 0, stream>>>(OUTI, P(0,0,4), nullptr,
                                                        N_ITEMS, D1, 1.0f, 1);
  bias_act<<<cdiv(N_USERS * D1, 256), 256, 0, stream>>>(OUTU, P(0,1,4), P(0,2,4),
                                                        N_USERS, D1, 0.5f, 1);

  // ---------------- Layer 2 (Din=256, H=1, C=128, out 128, mean) -----------
  float* out_u = (float*)d_out;                        // [N_USERS, 128]
  float* out_i = (float*)d_out + (size_t)N_USERS * D2; // [N_ITEMS, 128]
  fill_kernel<<<cdiv(N_USERS * D2, 256), 256, 0, stream>>>(out_u, 0.f, N_USERS * D2);
  fill_kernel<<<cdiv(N_ITEMS * D2, 256), 256, 0, stream>>>(out_i, 0.f, N_ITEMS * D2);

  run_relation<H2, C2>(OUTU, OUTI, N_USERS, N_ITEMS, e_rates,
                       P(1,0,0), P(1,0,1), P(1,0,2), P(1,0,3),
                       A, B, ALS, ALD, MB, DEN, EX, out_i, stream);
  run_relation<H2, C2>(OUTI, OUTU, N_ITEMS, N_USERS, e_rev,
                       P(1,1,0), P(1,1,1), P(1,1,2), P(1,1,3),
                       A, B, ALS, ALD, MB, DEN, EX, out_u, stream);
  run_relation<H2, C2>(OUTU, OUTU, N_USERS, N_USERS, e_fol,
                       P(1,2,0), P(1,2,1), P(1,2,2), P(1,2,3),
                       A, B, ALS, ALD, MB, DEN, EX, out_u, stream);

  bias_act<<<cdiv(N_ITEMS * D2, 256), 256, 0, stream>>>(out_i, P(1,0,4), nullptr,
                                                        N_ITEMS, D2, 1.0f, 0);
  bias_act<<<cdiv(N_USERS * D2, 256), 256, 0, stream>>>(out_u, P(1,1,4), P(1,2,4),
                                                        N_USERS, D2, 0.5f, 0);
}
